// SelfAttention_27058293965137
// MI455X (gfx1250) — matrix operand order repaired and checked by reading, not compiled, not run on hardware
//
#include <hip/hip_runtime.h>
#include <math.h>

// ---------------------------------------------------------------------------
// Types for CDNA5 WMMA (gfx1250, wave32)
// ---------------------------------------------------------------------------
typedef __bf16 v16bf __attribute__((ext_vector_type(16)));
typedef __bf16 v8bf  __attribute__((ext_vector_type(8)));
typedef float  v8f   __attribute__((ext_vector_type(8)));
typedef int    i32x4 __attribute__((vector_size(4 * sizeof(int))));

#define WMMA_BF16(a, b, c) \
    __builtin_amdgcn_wmma_f32_16x16x32_bf16(false, (a), false, (b), (short)0, (c), false, false)

// CDNA5 async global->LDS path (ASYNCcnt), guarded: falls back to VGPR staging
#if defined(__AMDGCN__) && __has_builtin(__builtin_amdgcn_global_load_async_to_lds_b128) && \
    __has_builtin(__builtin_amdgcn_s_wait_asynccnt)
#define HAVE_ASYNC_LDS 1
#else
#define HAVE_ASYNC_LDS 0
#endif

#if HAVE_ASYNC_LDS
typedef __attribute__((address_space(1))) i32x4 gvec_t;   // global int32x4
typedef __attribute__((address_space(3))) i32x4 lvec_t;   // LDS int32x4
static __device__ __forceinline__ void async_g2l_b128(const __bf16* g, __bf16* l) {
    __builtin_amdgcn_global_load_async_to_lds_b128((gvec_t*)g, (lvec_t*)l, 0, 0);
}
#endif

static __device__ __forceinline__ __bf16 f2bf(float f) {
    union { float f; unsigned u; } in; in.f = f;
    unsigned u = in.u + 0x7FFFu + ((in.u >> 16) & 1u);   // round-to-nearest-even
    union { unsigned short s; __bf16 b; } out; out.s = (unsigned short)(u >> 16);
    return out.b;
}

// Problem constants
#define Bdim 4
#define Ndim 2048
#define Cdim 1024
#define Hdim 16
#define Ddim 64
#define Mrows (Bdim * Ndim)   // 8192
#define BKP 48                // padded LDS K-row (32 used + 16 pad, keeps 32B align)

// ---------------------------------------------------------------------------
// fp32 -> bf16 convert (weights)
// ---------------------------------------------------------------------------
__global__ void cvt_f32_bf16(const float* __restrict__ src, __bf16* __restrict__ dst, int n) {
    for (int i = blockIdx.x * blockDim.x + threadIdx.x; i < n; i += gridDim.x * blockDim.x)
        dst[i] = f2bf(src[i]);
}

// ---------------------------------------------------------------------------
// LayerNorm (one 256-thread block per row of C=1024), bf16 output
// ---------------------------------------------------------------------------
__global__ __launch_bounds__(256) void ln_bf16(const float* __restrict__ x,
                                               const float* __restrict__ g,
                                               const float* __restrict__ b,
                                               __bf16* __restrict__ out) {
    const int row = blockIdx.x;
    const float* xr = x + (size_t)row * Cdim;
    float vals[4], s = 0.f, ss = 0.f;
#pragma unroll
    for (int i = 0; i < 4; i++) {
        float v = xr[threadIdx.x + i * 256];
        vals[i] = v; s += v; ss += v * v;
    }
#pragma unroll
    for (int off = 16; off; off >>= 1) {
        s  += __shfl_xor(s,  off, 32);
        ss += __shfl_xor(ss, off, 32);
    }
    __shared__ float sred[8], ssred[8];
    if ((threadIdx.x & 31) == 0) { sred[threadIdx.x >> 5] = s; ssred[threadIdx.x >> 5] = ss; }
    __syncthreads();
    float tot = 0.f, tot2 = 0.f;
#pragma unroll
    for (int i = 0; i < 8; i++) { tot += sred[i]; tot2 += ssred[i]; }
    const float mu = tot * (1.f / Cdim);
    const float var = tot2 * (1.f / Cdim) - mu * mu;
    const float rs = rsqrtf(var + 1e-5f);
#pragma unroll
    for (int i = 0; i < 4; i++) {
        int idx = threadIdx.x + i * 256;
        out[(size_t)row * Cdim + idx] = f2bf((vals[i] - mu) * rs * g[idx] + b[idx]);
    }
}

// ---------------------------------------------------------------------------
// Tiled bf16 WMMA GEMM: C[M,Nc] = A[M,K] @ W[K,Nc] (+bias, + epilogue)
// Block tile 128x128, K-step 32, double-buffered LDS; next-tile global loads
// in flight during WMMA compute (async-to-LDS for A when available).
// 8 waves: each computes 32x64 (2x4 WMMA tiles).
// ---------------------------------------------------------------------------
#define EPI_QKV   0
#define EPI_RESID 1
#define EPI_GELU  2

template <int EPI>
__global__ __launch_bounds__(256) void gemm_bf16(const __bf16* __restrict__ A,
                                                 const __bf16* __restrict__ W,
                                                 const float* __restrict__ bias,
                                                 int M, int K, int Nc,
                                                 float* __restrict__ outF,
                                                 __bf16* __restrict__ outB,
                                                 const float* __restrict__ resid,
                                                 __bf16* __restrict__ qkv, int strideQKV) {
    __shared__ __bf16 As[2][128][BKP];    // [buf][m][k]  24 KB
    __shared__ __bf16 BsT[2][128][BKP];   // [buf][n][k]  24 KB (K-major: b128 frag loads)

    const int tid  = threadIdx.x;
    const int bm   = blockIdx.y * 128, bn = blockIdx.x * 128;
    const int wave = tid >> 5, lane = tid & 31;
    const int lr = lane & 15, lh = lane >> 4;
    const int rowBase = (wave & 3) * 32;
    const int colBase = (wave >> 2) * 64;

    // global->LDS staging assignments
    const int amm = tid >> 1, akk = (tid & 1) * 16;          // A: 16 contig bf16
    const int bkk = tid >> 3, bnb = (tid & 7) * 16;          // W: 16 contig bf16

    v8f acc[2][4];
    const v8f vzero = {0.f, 0.f, 0.f, 0.f, 0.f, 0.f, 0.f, 0.f};
#pragma unroll
    for (int i = 0; i < 2; i++)
#pragma unroll
        for (int j = 0; j < 4; j++) acc[i][j] = vzero;

    // ---- prologue: stage tile 0 into buffer 0 ----
#if HAVE_ASYNC_LDS
    async_g2l_b128(&A[(size_t)(bm + amm) * K + akk],     &As[0][amm][akk]);
    async_g2l_b128(&A[(size_t)(bm + amm) * K + akk + 8], &As[0][amm][akk + 8]);
#else
    *(v16bf*)&As[0][amm][akk] = *(const v16bf*)&A[(size_t)(bm + amm) * K + akk];
#endif
    {
        v16bf wv = *(const v16bf*)&W[(size_t)bkk * Nc + bn + bnb];
#pragma unroll
        for (int j = 0; j < 16; j++) BsT[0][bnb + j][bkk] = wv[j];
    }
#if HAVE_ASYNC_LDS
    __builtin_amdgcn_s_wait_asynccnt(0);
#endif
    __syncthreads();

    for (int k0 = 0; k0 < K; k0 += 32) {
        const int cur = (k0 >> 5) & 1, nxt = cur ^ 1;
        const bool hasNext = (k0 + 32) < K;

        // issue next-tile global loads (in flight during compute)
        v16bf aNext, bNext;
        if (hasNext) {
#if HAVE_ASYNC_LDS
            async_g2l_b128(&A[(size_t)(bm + amm) * K + k0 + 32 + akk],     &As[nxt][amm][akk]);
            async_g2l_b128(&A[(size_t)(bm + amm) * K + k0 + 32 + akk + 8], &As[nxt][amm][akk + 8]);
#else
            aNext = *(const v16bf*)&A[(size_t)(bm + amm) * K + k0 + 32 + akk];
#endif
            bNext = *(const v16bf*)&W[(size_t)(k0 + 32 + bkk) * Nc + bn + bnb];
        }

        // A fragments (contiguous K order as in B: lane = row, halves split K)
        v16bf af[2];
#pragma unroll
        for (int im = 0; im < 2; im++) {
            const __bf16* ap = &As[cur][rowBase + im * 16 + lr][0];
            v8bf lo = *(const v8bf*)(ap + lh * 16);
            v8bf hi = *(const v8bf*)(ap + lh * 16 + 8);
            v16bf a;
#pragma unroll
            for (int i = 0; i < 8; i++) { a[i] = lo[i]; a[8 + i] = hi[i]; }
            af[im] = a;
        }
        // B fragments (lane = col, halves split K; contiguous in BsT)
        v16bf bfr[4];
#pragma unroll
        for (int jn = 0; jn < 4; jn++)
            bfr[jn] = *(const v16bf*)&BsT[cur][colBase + jn * 16 + lr][lh * 16];

#pragma unroll
        for (int im = 0; im < 2; im++)
#pragma unroll
            for (int jn = 0; jn < 4; jn++)
                acc[im][jn] = WMMA_BF16(af[im], bfr[jn], acc[im][jn]);

        // commit next tile to the other buffer
        if (hasNext) {
#if !HAVE_ASYNC_LDS
            *(v16bf*)&As[nxt][amm][akk] = aNext;
#endif
#pragma unroll
            for (int j = 0; j < 16; j++) BsT[nxt][bnb + j][bkk] = bNext[j];
        }
#if HAVE_ASYNC_LDS
        if (hasNext) __builtin_amdgcn_s_wait_asynccnt(0);
#endif
        __syncthreads();
    }

    // epilogue — D layout: lane holds col (lane&15), rows j + 8*(lane>>4)
#pragma unroll
    for (int im = 0; im < 2; im++)
#pragma unroll
        for (int jn = 0; jn < 4; jn++)
#pragma unroll
            for (int j = 0; j < 8; j++) {
                const int grow = bm + rowBase + im * 16 + lh * 8 + j;
                const int gcol = bn + colBase + jn * 16 + lr;
                float v = acc[im][jn][j] + bias[gcol];
                if (EPI == EPI_QKV) {
                    const int which = gcol >> 10;            // 0:q 1:k 2:v
                    const int cc = gcol & 1023;
                    const int head = cc >> 6, d = cc & 63;
                    const int bb = grow >> 11, n = grow & 2047;
                    if (which == 0) v *= 0.125f;             // D^-0.5 folded into q
                    qkv[(size_t)which * strideQKV +
                        ((((size_t)bb * Hdim + head) * Ndim + n) * Ddim + d)] = f2bf(v);
                } else if (EPI == EPI_RESID) {
                    const size_t o = (size_t)grow * Nc + gcol;
                    outF[o] = v + resid[o];
                } else {  // EPI_GELU (exact)
                    float gl = 0.5f * v * (1.f + erff(v * 0.70710678118654752f));
                    outB[(size_t)grow * Nc + gcol] = f2bf(gl);
                }
            }
}

// ---------------------------------------------------------------------------
// Flash attention: grid (N/128, B*H). 8 waves x 16 q-rows. K/V tiles of 64 keys
// staged in LDS with register prefetch of the next tile during compute.
// Q pre-scaled by D^-0.5. O written bf16 into [B,N,C] layout.
// ---------------------------------------------------------------------------
__global__ __launch_bounds__(256) void attn_fwd(const __bf16* __restrict__ Q,
                                                const __bf16* __restrict__ Kt,
                                                const __bf16* __restrict__ V,
                                                __bf16* __restrict__ O) {
    __shared__ __bf16 Ks[64][64];          // [key][d]   8 KB
    __shared__ __bf16 VtT[64][64];         // [d][key]   8 KB
    __shared__ __bf16 Pbuf[8][16][64];     // wave-private P staging  16 KB

    const int bh = blockIdx.y;
    const int b = bh >> 4, h = bh & 15;
    const int qbase = blockIdx.x * 128;
    const int tid = threadIdx.x, wave = tid >> 5, lane = tid & 31;
    const int lr = lane & 15, lh = lane >> 4;

    const __bf16* Qb = Q + (size_t)bh * Ndim * Ddim;
    const __bf16* Kb = Kt + (size_t)bh * Ndim * Ddim;
    const __bf16* Vb = V + (size_t)bh * Ndim * Ddim;

    // Q fragments (A layout), kept in registers for whole kernel
    const int qrow = qbase + wave * 16 + lr;
    v16bf qf[2];
#pragma unroll
    for (int f = 0; f < 2; f++) {
        const __bf16* qp = Qb + (size_t)qrow * Ddim + f * 32;
        v8bf lo = *(const v8bf*)(qp + lh * 16);
        v8bf hi = *(const v8bf*)(qp + lh * 16 + 8);
        v16bf a;
#pragma unroll
        for (int i = 0; i < 8; i++) { a[i] = lo[i]; a[8 + i] = hi[i]; }
        qf[f] = a;
    }

    const v8f vzero = {0.f, 0.f, 0.f, 0.f, 0.f, 0.f, 0.f, 0.f};
    v8f o[4];
#pragma unroll
    for (int jn = 0; jn < 4; jn++) o[jn] = vzero;
    float mrow[8], lrow[8];
#pragma unroll
    for (int j = 0; j < 8; j++) { mrow[j] = -3.0e38f; lrow[j] = 0.f; }

    // prefetch tile 0 into registers
    const int kkey = tid >> 2, kdb = (tid & 3) * 16;     // K staging
    const int vkey = tid & 63, vdb = (tid >> 6) * 16;    // V staging (transposed commit)
    v16bf kReg = *(const v16bf*)(Kb + (size_t)kkey * Ddim + kdb);
    v16bf vReg = *(const v16bf*)(Vb + (size_t)vkey * Ddim + vdb);

    for (int kt = 0; kt < Ndim; kt += 64) {
        // commit current tile to LDS
        *(v16bf*)&Ks[kkey][kdb] = kReg;
#pragma unroll
        for (int j = 0; j < 16; j++) VtT[vdb + j][vkey] = vReg[j];
        __syncthreads();

        // issue next-tile global loads (in flight during compute)
        if (kt + 64 < Ndim) {
            kReg = *(const v16bf*)(Kb + (size_t)(kt + 64 + kkey) * Ddim + kdb);
            vReg = *(const v16bf*)(Vb + (size_t)(kt + 64 + vkey) * Ddim + vdb);
        }

        // S = Q @ K^T  (16 rows x 64 keys per wave, 4 tiles x 2 WMMA)
        v8f s[4];
#pragma unroll
        for (int jn = 0; jn < 4; jn++) {
            const __bf16* kp = &Ks[jn * 16 + lr][0];
            v16bf bk0 = *(const v16bf*)(kp + lh * 16);
            v16bf bk1 = *(const v16bf*)(kp + 32 + lh * 16);
            v8f sz = vzero;
            sz = WMMA_BF16(qf[0], bk0, sz);
            sz = WMMA_BF16(qf[1], bk1, sz);
            s[jn] = sz;
        }

        // online softmax: row stats via 16-lane-half shuffles (matches C/D layout)
        float rm[8];
#pragma unroll
        for (int j = 0; j < 8; j++) {
            rm[j] = fmaxf(fmaxf(s[0][j], s[1][j]), fmaxf(s[2][j], s[3][j]));
#pragma unroll
            for (int off = 1; off < 16; off <<= 1)
                rm[j] = fmaxf(rm[j], __shfl_xor(rm[j], off, 16));
        }
        float alpha[8], rs[8];
#pragma unroll
        for (int j = 0; j < 8; j++) {
            float mn = fmaxf(mrow[j], rm[j]);
            alpha[j] = __expf(mrow[j] - mn);
            mrow[j] = mn;
            rs[j] = 0.f;
        }
#pragma unroll
        for (int jn = 0; jn < 4; jn++)
#pragma unroll
            for (int j = 0; j < 8; j++) {
                float p = __expf(s[jn][j] - mrow[j]);
                s[jn][j] = p;
                rs[j] += p;
            }
#pragma unroll
        for (int j = 0; j < 8; j++) {
#pragma unroll
            for (int off = 1; off < 16; off <<= 1)
                rs[j] += __shfl_xor(rs[j], off, 16);
            lrow[j] = lrow[j] * alpha[j] + rs[j];
        }
#pragma unroll
        for (int jn = 0; jn < 4; jn++)
#pragma unroll
            for (int j = 0; j < 8; j++) o[jn][j] *= alpha[j];

        // P: D layout -> A layout via wave-private LDS (in-order per wave)
#pragma unroll
        for (int jn = 0; jn < 4; jn++)
#pragma unroll
            for (int j = 0; j < 8; j++)
                Pbuf[wave][j + 8 * lh][jn * 16 + lr] = f2bf(s[jn][j]);

        v16bf pf[2];
#pragma unroll
        for (int f = 0; f < 2; f++) {
            const __bf16* pp = &Pbuf[wave][lr][0] + f * 32;
            v8bf lo = *(const v8bf*)(pp + lh * 16);
            v8bf hi = *(const v8bf*)(pp + lh * 16 + 8);
            v16bf a;
#pragma unroll
            for (int i = 0; i < 8; i++) { a[i] = lo[i]; a[8 + i] = hi[i]; }
            pf[f] = a;
        }

        // O += P @ V
#pragma unroll
        for (int jn = 0; jn < 4; jn++) {
            const __bf16* vp = &VtT[jn * 16 + lr][0];
            v16bf bv0 = *(const v16bf*)(vp + lh * 16);
            v16bf bv1 = *(const v16bf*)(vp + 32 + lh * 16);
            o[jn] = WMMA_BF16(pf[0], bv0, o[jn]);
            o[jn] = WMMA_BF16(pf[1], bv1, o[jn]);
        }
        __syncthreads();
    }

    float inv[8];
#pragma unroll
    for (int j = 0; j < 8; j++) inv[j] = (lrow[j] > 0.f) ? 1.f / lrow[j] : 0.f;
#pragma unroll
    for (int jn = 0; jn < 4; jn++)
#pragma unroll
        for (int j = 0; j < 8; j++) {
            const int n = qbase + wave * 16 + 8 * lh + j;
            const int d = jn * 16 + lr;
            O[((size_t)b * Ndim + n) * Cdim + h * Ddim + d] = f2bf(o[jn][j] * inv[j]);
        }
}

// ---------------------------------------------------------------------------
// Orchestration
// ---------------------------------------------------------------------------
extern "C" void kernel_launch(void* const* d_in, const int* in_sizes, int n_in,
                              void* d_out, int out_size, void* d_ws, size_t ws_size,
                              hipStream_t stream) {
    const float* x     = (const float*)d_in[0];
    const float* Wqkv  = (const float*)d_in[1];
    const float* bqkv  = (const float*)d_in[2];
    const float* Wproj = (const float*)d_in[3];
    const float* bproj = (const float*)d_in[4];
    const float* g1    = (const float*)d_in[5];
    const float* b1    = (const float*)d_in[6];
    const float* g2    = (const float*)d_in[7];
    const float* b2    = (const float*)d_in[8];
    const float* W1    = (const float*)d_in[9];
    const float* bf1   = (const float*)d_in[10];
    const float* W2    = (const float*)d_in[11];
    const float* bf2   = (const float*)d_in[12];
    float* out = (float*)d_out;

    char* ws = (char*)d_ws;
    const size_t SZA = (size_t)Mrows * Cdim * 2;          // 16 MB per bf16 activation
    __bf16* hA    = (__bf16*)(ws);                        // LN1 out (reused for LN2 out)
    __bf16* q     = (__bf16*)(ws + SZA);                  // [B,H,N,D]
    __bf16* qkvb  = q;                                    // scatter base
    __bf16* Ob    = (__bf16*)(ws + 4 * SZA);              // attn out [B,N,C]
    __bf16* ffn1  = (__bf16*)(ws + SZA);                  // aliases q/k/v/O (dead by then)
    float*  x1    = (float*)(ws + 5 * SZA);               // post-attn residual fp32
    __bf16* Wqkvb = (__bf16*)(ws + 5 * SZA + (size_t)Mrows * Cdim * 4);
    __bf16* Wprojb= Wqkvb + (size_t)Cdim * 3 * Cdim;
    __bf16* W1b   = Wprojb + (size_t)Cdim * Cdim;
    __bf16* W2b   = W1b + (size_t)Cdim * 4 * Cdim;

    // 1) weights -> bf16
    cvt_f32_bf16<<<1024, 256, 0, stream>>>(Wqkv,  Wqkvb,  Cdim * 3 * Cdim);
    cvt_f32_bf16<<<1024, 256, 0, stream>>>(Wproj, Wprojb, Cdim * Cdim);
    cvt_f32_bf16<<<1024, 256, 0, stream>>>(W1,    W1b,    Cdim * 4 * Cdim);
    cvt_f32_bf16<<<1024, 256, 0, stream>>>(W2,    W2b,    4 * Cdim * Cdim);

    // 2) LN1
    ln_bf16<<<Mrows, 256, 0, stream>>>(x, g1, b1, hA);

    // 3) QKV GEMM [8192,1024] x [1024,3072] -> q/k/v [B,H,N,D] bf16 (q pre-scaled)
    gemm_bf16<EPI_QKV><<<dim3(3 * Cdim / 128, Mrows / 128), 256, 0, stream>>>(
        hA, Wqkvb, bqkv, Mrows, Cdim, 3 * Cdim,
        nullptr, nullptr, nullptr, qkvb, Bdim * Hdim * Ndim * Ddim);

    // 4) attention
    attn_fwd<<<dim3(Ndim / 128, Bdim * Hdim), 256, 0, stream>>>(
        q, q + (size_t)Bdim * Hdim * Ndim * Ddim, q + 2 * (size_t)Bdim * Hdim * Ndim * Ddim, Ob);

    // 5) proj + residual(x) -> x1 fp32
    gemm_bf16<EPI_RESID><<<dim3(Cdim / 128, Mrows / 128), 256, 0, stream>>>(
        Ob, Wprojb, bproj, Mrows, Cdim, Cdim, x1, nullptr, x, nullptr, 0);

    // 6) LN2 (reuse hA)
    ln_bf16<<<Mrows, 256, 0, stream>>>(x1, g2, b2, hA);

    // 7) FFN1 + exact GELU -> bf16 (aliases dead q/k/v/O region)
    gemm_bf16<EPI_GELU><<<dim3(4 * Cdim / 128, Mrows / 128), 256, 0, stream>>>(
        hA, W1b, bf1, Mrows, Cdim, 4 * Cdim, nullptr, ffn1, nullptr, nullptr, 0);

    // 8) FFN2 + residual(x1) -> d_out fp32
    gemm_bf16<EPI_RESID><<<dim3(Cdim / 128, Mrows / 128), 256, 0, stream>>>(
        ffn1, W2b, bf2, Mrows, 4 * Cdim, Cdim, out, nullptr, x1, nullptr, 0);

    (void)in_sizes; (void)n_in; (void)out_size; (void)ws_size;
}
